// Transformer_48696339202580
// MI455X (gfx1250) — compile-verified, matrix operand order checked by reading, not run on hardware
//
#include <hip/hip_runtime.h>

// ---------------------------------------------------------------------------
// Graph-transformer forward for MI455X (gfx1250, wave32, WMMA).
// All dense projections run as bf16 WMMA GEMMs (v_wmma_f32_16x16x32_bf16,
// fp32 accumulate) with compile-time K so the K-loop fully unrolls and the
// scheduler can software-pipeline loads across WMMAs. Output dtype and bias
// presence are compile-time template flags (no runtime branches in epilogue).
// Edge attention: wave-per-edge gathers + shuffle reductions; segment-sum via
// fp32 global atomics.
// ---------------------------------------------------------------------------

#define DEV_N 50000
#define DEV_E 800000

typedef __attribute__((ext_vector_type(8)))  __bf16 v8bf;
typedef __attribute__((ext_vector_type(16))) __bf16 v16bf;
typedef __attribute__((ext_vector_type(4)))  __bf16 v4bf;
typedef __attribute__((ext_vector_type(8)))  float  v8f;

// ---------------- fragment load (A-layout for 16-bit 16x32) ----------------
// lanes 0-15: row M = lane,  K runs [0..7] and [16..23]
// lanes 16-31: row M = lane-16, K runs [8..15] and [24..31]
__device__ __forceinline__ v16bf ld_frag16(const __bf16* __restrict__ base, int off1) {
    v8bf lo = *(const v8bf*)(base + off1);
    v8bf hi = *(const v8bf*)(base + off1 + 16);
    return __builtin_shufflevector(lo, hi, 0,1,2,3,4,5,6,7,8,9,10,11,12,13,14,15);
}

// ---------------- WMMA GEMM: out = act(A[M,K](bf16) @ Bt^T + bias) ---------
// Bt is the pre-transposed weight: [Nd rows, ldb(>=K) cols] bf16.
// Each wave computes a 16 x (16*NT) tile; K = 32*KT is compile-time.
// ACT: 0=none,1=relu,2=selu. OUTB: 0 -> fp32 out, 1 -> bf16 out. HASB: bias.
template<int NT, int ACT, int KT, int OUTB, bool HASB>
__global__ __launch_bounds__(256) void k_gemm(
    const __bf16* __restrict__ A, int lda,
    const __bf16* __restrict__ Bt, int ldb,
    const float*  __restrict__ bias,
    void*         __restrict__ outP,
    int M, int Nd)
{
    const int lane = threadIdx.x & 31;
    const int wave = threadIdx.x >> 5;
    const int ntn  = Nd / (16 * NT);
    const long gw  = (long)blockIdx.x * 8 + wave;
    const long tot = (long)(M / 16) * ntn;
    if (gw >= tot) return;                       // whole-wave uniform exit
    const int tm   = (int)(gw / ntn) * 16;
    const int tn   = (int)(gw % ntn) * (16 * NT);
    const int half = lane >> 4;
    const int lidx = lane & 15;
    const int off1 = half * 8;

    const __bf16* __restrict__ arow = A + (long)(tm + lidx) * lda;
    const __bf16* __restrict__ brow[NT];
    #pragma unroll
    for (int t = 0; t < NT; ++t)
        brow[t] = Bt + (long)(tn + t * 16 + lidx) * ldb;

    v8f c[NT];
    #pragma unroll
    for (int t = 0; t < NT; ++t) c[t] = (v8f){0.f,0.f,0.f,0.f,0.f,0.f,0.f,0.f};

    #pragma unroll
    for (int s = 0; s < KT; ++s) {
        const int kk = s * 32;
        // issue all loads of this step before any WMMA consumes them
        v16bf a = ld_frag16(arow + kk, off1);
        v16bf b[NT];
        #pragma unroll
        for (int t = 0; t < NT; ++t) b[t] = ld_frag16(brow[t] + kk, off1);
        #pragma unroll
        for (int t = 0; t < NT; ++t)
            c[t] = __builtin_amdgcn_wmma_f32_16x16x32_bf16(
                       false, a, false, b[t], (short)0, c[t], false, false);
    }

    #pragma unroll
    for (int t = 0; t < NT; ++t) {
        const int n  = tn + t * 16 + lidx;
        const float bv = HASB ? bias[n] : 0.f;
        #pragma unroll
        for (int r = 0; r < 8; ++r) {
            const int m = tm + r + half * 8;
            float v = c[t][r] + bv;
            if (ACT == 1) v = fmaxf(v, 0.f);
            if (ACT == 2) v = (v > 0.f) ? 1.0507009873554805f * v
                                        : 1.7580993408473766f * (__expf(v) - 1.f);
            const long o = (long)m * Nd + n;
            if (OUTB == 0) ((float*)outP)[o]  = v;
            else           ((__bf16*)outP)[o] = (__bf16)v;
        }
    }
}

// ---------------- fp32 -> bf16 with column padding --------------------------
__global__ void k_cvt_pad(const float* __restrict__ src, __bf16* __restrict__ dst,
                          long rows, int cs, int cd)
{
    long i = (long)blockIdx.x * blockDim.x + threadIdx.x;
    if (i >= rows * cd) return;
    long r = i / cd; int c = (int)(i % cd);
    dst[i] = (c < cs) ? (__bf16)src[r * cs + c] : (__bf16)0.f;
}

// ---------------- weight transpose fp32[K,Nd] -> bf16[Nd,Kpad] -------------
__global__ void k_wt(const float* __restrict__ W, __bf16* __restrict__ Wt,
                     int K, int Nd, int Kpad)
{
    long i = (long)blockIdx.x * blockDim.x + threadIdx.x;
    if (i >= (long)Nd * Kpad) return;
    int n = (int)(i / Kpad), k = (int)(i % Kpad);
    Wt[i] = (k < K) ? (__bf16)W[(long)k * Nd + n] : (__bf16)0.f;
}

// ---------------- fused  (emb @ headsum(pd.W))  composition -----------------
// pd.sum(-1)[e,h] = de[e,:fi] @ Wc[:,h] + bc[h], Wc = W1 @ Wsum(W2), h<8.
// Output WcT: [16, 32] bf16 transposed+padded; bc: [16] fp32.
__global__ void k_wcomb(const float* __restrict__ W1, const float* __restrict__ b1,
                        const float* __restrict__ W2, const float* __restrict__ b2,
                        int fi, __bf16* __restrict__ WcT, float* __restrict__ bc)
{
    int t = threadIdx.x;            // 512 threads, grid 1
    int h = t >> 5, j = t & 31;
    float acc = 0.f;
    if (h < 8 && j < fi) {
        for (int k = 0; k < 128; ++k) {
            float ws = 0.f;
            #pragma unroll
            for (int d = 0; d < 16; ++d) ws += W2[k * 128 + h * 16 + d];
            acc += W1[j * 128 + k] * ws;
        }
    }
    WcT[h * 32 + j] = (__bf16)acc;
    if (j == 0) {
        float bb = 0.f;
        if (h < 8) {
            for (int k = 0; k < 128; ++k) {
                float ws = 0.f;
                for (int d = 0; d < 16; ++d) ws += W2[k * 128 + h * 16 + d];
                bb += b1[k] * ws;
            }
            for (int d = 0; d < 16; ++d) bb += b2[h * 16 + d];
        }
        bc[h] = bb;
    }
}

// ---------------- per-edge attention scores (wave per edge) -----------------
__global__ __launch_bounds__(256) void k_edge_score(
    const float* __restrict__ Qf, const float* __restrict__ Kf,
    const float* __restrict__ pd, const float* __restrict__ pm,
    const int* __restrict__ src, const int* __restrict__ dst,
    float* __restrict__ score, int E)
{
    long e = (long)blockIdx.x * 8 + (threadIdx.x >> 5);
    if (e >= E) return;
    int lane = threadIdx.x & 31;
    int s = src[e], d = dst[e];
    float4 kv = ((const float4*)(Kf + (long)s * 128))[lane];
    float4 qv = ((const float4*)(Qf + (long)d * 128))[lane];
    float p = kv.x*qv.x + kv.y*qv.y + kv.z*qv.z + kv.w*qv.w;
    p += __shfl_xor(p, 1, 32);
    p += __shfl_xor(p, 2, 32);      // lanes 4h hold head-h dot16
    int h = lane >> 2;
    if ((lane & 3) == 0) {
        // s = DH*score0 + pd + pm;  DH*dot/sqrt(DH) = 4*dot for DH=16
        float sv = 4.f * p + pd[e * 16 + h] + pm[e * 16 + h];
        sv = fminf(10.f, fmaxf(-10.f, sv));
        score[e * 8 + h] = __expf(sv) * 0.5f;
    }
}

// ---------------- zero fill -------------------------------------------------
__global__ void k_zero(float* __restrict__ p, long n) {
    long i = (long)blockIdx.x * blockDim.x + threadIdx.x;
    if (i < n) p[i] = 0.f;
}

// ---------------- segment-sum scatter (atomics) -----------------------------
__global__ void k_scatter(const float* __restrict__ Vf, const float* __restrict__ score,
                          const int* __restrict__ src, const int* __restrict__ dst,
                          float* __restrict__ wV, float* __restrict__ z, int E)
{
    long i = (long)blockIdx.x * blockDim.x + threadIdx.x;
    if (i >= (long)E * 128) return;
    long e = i >> 7; int c = (int)(i & 127);
    float sc = score[e * 8 + (c >> 4)];
    float v  = Vf[(long)src[e] * 128 + c] * sc;
    __hip_atomic_fetch_add(&wV[(long)dst[e] * 128 + c], v,
                           __ATOMIC_RELAXED, __HIP_MEMORY_SCOPE_AGENT);
    if (c < 8)
        __hip_atomic_fetch_add(&z[(long)dst[e] * 8 + c], score[e * 8 + c],
                               __ATOMIC_RELAXED, __HIP_MEMORY_SCOPE_AGENT);
}

// ---------------- attn normalize + add i(I), emit bf16 ----------------------
__global__ void k_combine(const float* __restrict__ wV, const float* __restrict__ z,
                          const float* __restrict__ iI, __bf16* __restrict__ preOb, long n)
{
    long i = (long)blockIdx.x * blockDim.x + threadIdx.x;
    if (i >= n) return;
    long r = i >> 7; int c = (int)(i & 127);
    float a = wV[i] / (z[r * 8 + (c >> 4)] + 1e-6f) + iI[i];
    preOb[i] = (__bf16)a;
}

// ---------------- residual + layernorm (wave per 128-wide row) --------------
__global__ __launch_bounds__(256) void k_add_ln(
    const float* __restrict__ a, const float* __restrict__ b,
    const float* __restrict__ g, const float* __restrict__ be,
    float* __restrict__ outF, __bf16* __restrict__ outB, int rows)
{
    long row = (long)blockIdx.x * 8 + (threadIdx.x >> 5);
    if (row >= rows) return;
    int lane = threadIdx.x & 31;
    float4 av = ((const float4*)(a + row * 128))[lane];
    float4 bv = ((const float4*)(b + row * 128))[lane];
    float x0 = av.x + bv.x, x1 = av.y + bv.y, x2 = av.z + bv.z, x3 = av.w + bv.w;
    float s = x0 + x1 + x2 + x3;
    #pragma unroll
    for (int m = 1; m < 32; m <<= 1) s += __shfl_xor(s, m, 32);
    float mu = s * (1.f / 128.f);
    float d0 = x0 - mu, d1 = x1 - mu, d2 = x2 - mu, d3 = x3 - mu;
    float q = d0*d0 + d1*d1 + d2*d2 + d3*d3;
    #pragma unroll
    for (int m = 1; m < 32; m <<= 1) q += __shfl_xor(q, m, 32);
    float rstd = rsqrtf(q * (1.f / 128.f) + 1e-5f);
    float4 gv  = ((const float4*)g)[lane];
    float4 bev = ((const float4*)be)[lane];
    float y0 = d0 * rstd * gv.x + bev.x;
    float y1 = d1 * rstd * gv.y + bev.y;
    float y2 = d2 * rstd * gv.z + bev.z;
    float y3 = d3 * rstd * gv.w + bev.w;
    ((float4*)(outF + row * 128))[lane] = make_float4(y0, y1, y2, y3);
    v4bf ob = {(__bf16)y0, (__bf16)y1, (__bf16)y2, (__bf16)y3};
    ((v4bf*)(outB + row * 128))[lane] = ob;
}

// ---------------- add + convert (embedding merge) ---------------------------
__global__ void k_add_cvt(const float* __restrict__ a, const float* __restrict__ b,
                          float* __restrict__ oF, __bf16* __restrict__ oB, long n)
{
    long i = (long)blockIdx.x * blockDim.x + threadIdx.x;
    if (i >= n) return;
    float v = a[i] + b[i];
    oF[i] = v;
    oB[i] = (__bf16)v;
}

// ===========================================================================
// Host-side orchestration
// ===========================================================================
static void gemm_launch(hipStream_t st, const __bf16* A, int lda, const __bf16* Bt,
                        int ldb, const float* bias, float* oF, __bf16* oB,
                        int M, int K, int Nd, int nt, int act)
{
    long ntn = Nd / (16 * nt);
    long waves = (long)(M / 16) * ntn;
    dim3 g((unsigned)((waves + 7) / 8)), bl(256);
    // Used combos only (exactly one of oF/oB non-null at every call site):
    if (nt == 1) {                                   // K=32, act=0, f32 out, bias
        k_gemm<1,0,1,0,true><<<g,bl,0,st>>>(A,lda,Bt,ldb,bias,oF,M,Nd);
    } else if (K == 32) {                            // act=0, f32 out, bias
        k_gemm<4,0,1,0,true><<<g,bl,0,st>>>(A,lda,Bt,ldb,bias,oF,M,Nd);
    } else if (K == 256) {                           // act=0, f32 out, bias
        k_gemm<4,0,8,0,true><<<g,bl,0,st>>>(A,lda,Bt,ldb,bias,oF,M,Nd);
    } else if (act == 1) {                           // K=128, relu, bf16 out, bias
        k_gemm<4,1,4,1,true><<<g,bl,0,st>>>(A,lda,Bt,ldb,bias,oB,M,Nd);
    } else if (act == 2) {                           // K=128, selu, bf16 out, bias
        k_gemm<4,2,4,1,true><<<g,bl,0,st>>>(A,lda,Bt,ldb,bias,oB,M,Nd);
    } else if (bias == nullptr) {                    // K=128, none, f32 out, no bias
        k_gemm<4,0,4,0,false><<<g,bl,0,st>>>(A,lda,Bt,ldb,bias,oF,M,Nd);
    } else {                                         // K=128, none, f32 out, bias
        k_gemm<4,0,4,0,true><<<g,bl,0,st>>>(A,lda,Bt,ldb,bias,oF,M,Nd);
    }
}

extern "C" void kernel_launch(void* const* d_in, const int* in_sizes, int n_in,
                              void* d_out, int out_size, void* d_ws, size_t ws_size,
                              hipStream_t stream)
{
    (void)in_sizes; (void)n_in; (void)out_size; (void)ws_size;
    const int N = DEV_N, E = DEV_E;

    auto F = [&](int i) { return (const float*)d_in[i]; };
    const float* X  = F(0);
    const float* PE = F(1);
    const float* I  = F(2);
    const float* m  = F(3);
    const float* de = F(4);
    const int* src  = (const int*)d_in[5];
    const int* dst  = (const int*)d_in[6];
    // params flattened (jax pytree sorted-key order):
    // emb_de{W,b}=7,8  emb_h{W}=9  emb_m{W,b}=10,11  lap{W,b}=12,13
    // layers[l] base 14+22l, order: K.W,K.b,O.W,O.b,Q.W,Q.b,V.W,V.b,
    //   f1.W,f1.b,f2.W,f2.b,i.W,i.b,ln1.b,ln1.g,ln2.b,ln2.g,pd.W,pd.b,pm.W,pm.b
    // mlp1{W,b}=80,81  mlp2{W,b}=82,83
    const int LB = 14, LS = 22;
    auto LP = [&](int l, int k) { return F(LB + LS * l + k); };

    // ---- workspace arena -------------------------------------------------
    size_t off = 0;
    auto alloc = [&](size_t bytes) {
        void* p = (char*)d_ws + off;
        off = (off + bytes + 255) & ~(size_t)255;
        return p;
    };
    __bf16* Xb    = (__bf16*)alloc((size_t)N * 128 * 2);
    __bf16* PEb   = (__bf16*)alloc((size_t)N * 32 * 2);
    __bf16* Ib    = (__bf16*)alloc((size_t)N * 32 * 2);
    __bf16* mb    = (__bf16*)alloc((size_t)E * 32 * 2);
    __bf16* deb   = (__bf16*)alloc((size_t)E * 32 * 2);
    __bf16* hb    = (__bf16*)alloc((size_t)N * 128 * 2);
    __bf16* preOb = (__bf16*)alloc((size_t)N * 128 * 2);
    __bf16* ffnb  = (__bf16*)alloc((size_t)N * 256 * 2);
    __bf16* mlp1b = (__bf16*)alloc((size_t)N * 128 * 2);
    __bf16* Wtmp  = (__bf16*)alloc((size_t)256 * 256 * 2);
    __bf16* WcTd  = (__bf16*)alloc((size_t)16 * 32 * 2);
    __bf16* WcTm  = (__bf16*)alloc((size_t)16 * 32 * 2);
    float*  bcd   = (float*)alloc(16 * 4);
    float*  bcm   = (float*)alloc(16 * 4);
    float*  Qf    = (float*)alloc((size_t)N * 128 * 4);
    float*  Kf    = (float*)alloc((size_t)N * 128 * 4);
    float*  Vf    = (float*)alloc((size_t)N * 128 * 4);
    float*  iIf   = (float*)alloc((size_t)N * 128 * 4);
    float*  h2f   = (float*)alloc((size_t)N * 128 * 4);
    float*  pds   = (float*)alloc((size_t)E * 16 * 4);
    float*  pms   = (float*)alloc((size_t)E * 16 * 4);
    float*  score = (float*)alloc((size_t)E * 8 * 4);
    float*  wV    = (float*)alloc((size_t)N * 128 * 4);
    float*  z     = (float*)alloc((size_t)N * 8 * 4);

    float* hf   = (float*)d_out;            // final h lives in d_out[0 : N*128)
    float* xhat = (float*)d_out + (size_t)N * 128;

    auto cvt = [&](const float* s, __bf16* d, long rows, int cs, int cd) {
        long n = rows * cd;
        k_cvt_pad<<<(unsigned)((n + 255) / 256), 256, 0, stream>>>(s, d, rows, cs, cd);
    };
    auto wt = [&](const float* W, int K, int Nd, int Kpad) {
        long n = (long)Nd * Kpad;
        k_wt<<<(unsigned)((n + 255) / 256), 256, 0, stream>>>(W, Wtmp, K, Nd, Kpad);
    };
    auto zero = [&](float* p, long n) {
        k_zero<<<(unsigned)((n + 255) / 256), 256, 0, stream>>>(p, n);
    };

    // ---- stage inputs to bf16 (pad K to 32 where needed) ------------------
    cvt(X,  Xb,  N, 128, 128);
    cvt(PE, PEb, N, 16, 32);
    cvt(I,  Ib,  N, 32, 32);
    cvt(m,  mb,  E, 8, 32);
    cvt(de, deb, E, 32, 32);

    // ---- embedding: h0 = X@emb_h.W + lap(PE) ------------------------------
    wt(F(9), 128, 128, 128);
    gemm_launch(stream, Xb, 128, Wtmp, 128, nullptr, Qf, nullptr, N, 128, 128, 4, 0);
    wt(F(12), 16, 128, 32);
    gemm_launch(stream, PEb, 32, Wtmp, 32, F(13), Kf, nullptr, N, 32, 128, 4, 0);
    k_add_cvt<<<(unsigned)(((long)N * 128 + 255) / 256), 256, 0, stream>>>(
        Qf, Kf, hf, hb, (long)N * 128);

    // ---- transformer layers ----------------------------------------------
    for (int l = 0; l < 3; ++l) {
        // Q, K, V projections (128x128)
        wt(LP(l, 4), 128, 128, 128);
        gemm_launch(stream, hb, 128, Wtmp, 128, LP(l, 5), Qf, nullptr, N, 128, 128, 4, 0);
        wt(LP(l, 0), 128, 128, 128);
        gemm_launch(stream, hb, 128, Wtmp, 128, LP(l, 1), Kf, nullptr, N, 128, 128, 4, 0);
        wt(LP(l, 6), 128, 128, 128);
        gemm_launch(stream, hb, 128, Wtmp, 128, LP(l, 7), Vf, nullptr, N, 128, 128, 4, 0);

        // fused head-sum projections: pd from de, pm from m  (E x 32 x 16)
        k_wcomb<<<1, 512, 0, stream>>>(F(7), F(8), LP(l, 18), LP(l, 19), 32, WcTd, bcd);
        gemm_launch(stream, deb, 32, WcTd, 32, bcd, pds, nullptr, E, 32, 16, 1, 0);
        k_wcomb<<<1, 512, 0, stream>>>(F(10), F(11), LP(l, 20), LP(l, 21), 8, WcTm, bcm);
        gemm_launch(stream, mb, 32, WcTm, 32, bcm, pms, nullptr, E, 32, 16, 1, 0);

        // edge scores + segment sums
        k_edge_score<<<(unsigned)((E + 7) / 8), 256, 0, stream>>>(
            Qf, Kf, pds, pms, src, dst, score, E);
        zero(wV, (long)N * 128);
        zero(z, (long)N * 8);
        k_scatter<<<(unsigned)(((long)E * 128 + 255) / 256), 256, 0, stream>>>(
            Vf, score, src, dst, wV, z, E);

        // i(I) projection, combine, O projection
        wt(LP(l, 12), 32, 128, 32);
        gemm_launch(stream, Ib, 32, Wtmp, 32, LP(l, 13), iIf, nullptr, N, 32, 128, 4, 0);
        k_combine<<<(unsigned)(((long)N * 128 + 255) / 256), 256, 0, stream>>>(
            wV, z, iIf, preOb, (long)N * 128);
        wt(LP(l, 2), 128, 128, 128);
        gemm_launch(stream, preOb, 128, Wtmp, 128, LP(l, 3), h2f, nullptr, N, 128, 128, 4, 0);
        k_add_ln<<<(unsigned)((N + 7) / 8), 256, 0, stream>>>(
            hf, h2f, LP(l, 15), LP(l, 14), hf, hb, N);

        // FFN: relu(h@f1)@f2, residual + LN
        wt(LP(l, 8), 128, 256, 128);
        gemm_launch(stream, hb, 128, Wtmp, 128, LP(l, 9), nullptr, ffnb, N, 128, 256, 4, 1);
        wt(LP(l, 10), 256, 128, 256);
        gemm_launch(stream, ffnb, 256, Wtmp, 256, LP(l, 11), h2f, nullptr, N, 256, 128, 4, 0);
        k_add_ln<<<(unsigned)((N + 7) / 8), 256, 0, stream>>>(
            hf, h2f, LP(l, 17), LP(l, 16), hf, hb, N);
    }

    // ---- reconstruction MLP: x_hat = selu(h@mlp1)@mlp2 --------------------
    wt(F(80), 128, 128, 128);
    gemm_launch(stream, hb, 128, Wtmp, 128, F(81), nullptr, mlp1b, N, 128, 128, 4, 2);
    wt(F(82), 128, 128, 128);
    gemm_launch(stream, mlp1b, 128, Wtmp, 128, F(83), xhat, nullptr, N, 128, 128, 4, 0);
}
